// ShidoTreeLSTMLayer_50354196578421
// MI455X (gfx1250) — matrix operand order repaired and checked by reading, not compiled, not run on hardware
//
#include <hip/hip_runtime.h>
#include <stdint.h>
#include <stddef.h>

#define LVLS 4
#define NN   2048
#define KK   8
#define DIN  256
#define DD   128
#define G4   512   // 4*DD

typedef __bf16 v16bf __attribute__((ext_vector_type(16)));
typedef __bf16 v8bf  __attribute__((ext_vector_type(8)));
typedef float  v8f   __attribute__((ext_vector_type(8)));

// hardware f32 -> bf16 conversion (backend selects v_cvt path on gfx1250)
static __device__ __forceinline__ __bf16 f2bf(float f) { return (__bf16)f; }

static __device__ __forceinline__ v16bf cat16(v8bf a, v8bf b) {
  v16bf r;
  #pragma unroll
  for (int i = 0; i < 8; ++i) { r[i] = a[i]; r[i + 8] = b[i]; }
  return r;
}

static __device__ __forceinline__ v8f wmma_bf(v16bf a, v16bf b, v8f c) {
  // D = A(16x32 bf16) * B(32x16 bf16) + C(16x16 f32)
  return __builtin_amdgcn_wmma_f32_16x16x32_bf16(false, a, false, b, (short)0, c,
                                                 false, false);
}

static __device__ __forceinline__ float rcpf_(float x) {
  return __builtin_amdgcn_rcpf(x);
}
static __device__ __forceinline__ float sigmoidf_(float x) {
  return rcpf_(1.0f + __expf(-x));
}
static __device__ __forceinline__ float tanhf_(float x) {
  float e = __expf(-2.0f * x);
  return (1.0f - e) * rcpf_(1.0f + e);
}

// low 32 bits of a generic pointer to __shared__ = wave-relative LDS byte offset
static __device__ __forceinline__ uint32_t lds_off(const void* p) {
  return (uint32_t)(uintptr_t)p;
}
// async global->LDS 16B copy per lane (ASYNCcnt-tracked)
static __device__ __forceinline__ void async_copy16(uint32_t ldsb, const void* g) {
  asm volatile("global_load_async_to_lds_b128 %0, %1, off"
               :: "v"(ldsb), "v"(g) : "memory");
}

// fragment geometry
#define SM_WFRAG_ELEMS  (8 * 32 * 32 * 16)  // K=256 (8 ktiles) x 32 ntiles
#define SM_FCFRAG_ELEMS (4 * 8 * 32 * 16)   // K=128 (4 ktiles) x 8 ntiles
#define SM_HST_ELEMS    (2 * 16 * DD)
#define SM_YL_ELEMS     (2 * 16 * DD)

// ---------------------------------------------------------------------------
// One-time packers: reorder weights into WMMA B-fragment order (bf16) so the
// per-block LDS fill is a straight async DMA, and convert x to bf16.
// ---------------------------------------------------------------------------
__global__ void k_pack_wcat(const float* __restrict__ uWih,
                            const float* __restrict__ uWhh,
                            __bf16* __restrict__ wpk) {
  int i = blockIdx.x * blockDim.x + threadIdx.x;
  if (i >= 8 * SM_WFRAG_ELEMS) return;
  int gd = i >> 17;
  int r = i & (SM_WFRAG_ELEMS - 1);
  int e = r & 15, ln = (r >> 4) & 31, nt = (r >> 9) & 31, kt = r >> 14;
  int n = nt * 16 + (ln & 15);            // output column 0..511
  int kk = kt * 32 + (e >> 3) * 16 + (ln >> 4) * 8 + (e & 7);  // K 0..255
  const float* Wih = uWih + (size_t)gd * G4 * DD;
  const float* Whh = uWhh + (size_t)gd * G4 * DD;
  float v = (kk < DD) ? Wih[(size_t)n * DD + kk]
                      : Whh[(size_t)n * DD + (kk - DD)];
  wpk[i] = f2bf(v);
}

__global__ void k_pack_fc(const float* __restrict__ ufc, __bf16* __restrict__ fcpk) {
  int i = blockIdx.x * blockDim.x + threadIdx.x;
  if (i >= 8 * SM_FCFRAG_ELEMS) return;
  int gd = i >> 14;
  int r = i & (SM_FCFRAG_ELEMS - 1);
  int gate = gd >> 1, dir = gd & 1;
  int e = r & 15, ln = (r >> 4) & 31, nt = (r >> 9) & 7, kt = r >> 12;
  int n = nt * 16 + (ln & 15);            // output D col
  int kk = kt * 32 + (e >> 3) * 16 + (ln >> 4) * 8 + (e & 7);  // within-dir concat dim
  const float* fcp = ufc + (size_t)gate * (2 * DD) * DD;
  fcpk[i] = f2bf(fcp[(size_t)(dir * DD + kk) * DD + n]);
}

__global__ void k_pack_ww(const float* __restrict__ Ww, __bf16* __restrict__ wwpk) {
  int i = blockIdx.x * blockDim.x + threadIdx.x;
  if (i >= SM_WFRAG_ELEMS) return;
  int e = i & 15, ln = (i >> 4) & 31, nt = (i >> 9) & 31, kt = i >> 14;
  int n = nt * 16 + (ln & 15);            // col 0..511
  int kk = kt * 32 + (e >> 3) * 16 + (ln >> 4) * 8 + (e & 7);  // K 0..255
  wwpk[i] = f2bf(Ww[(size_t)kk * G4 + n]);
}

__global__ void k_pack_x(const float* __restrict__ x, __bf16* __restrict__ xbf) {
  int i = blockIdx.x * blockDim.x + threadIdx.x;
  if (i < LVLS * NN * DIN) xbf[i] = f2bf(x[i]);
}

// ---------------------------------------------------------------------------
// Gather child h/c per node; one block per node (scalar index loads).
// ---------------------------------------------------------------------------
__global__ void k_gather(const int* __restrict__ indices,
                         const float* __restrict__ prevh,
                         const float* __restrict__ prevc,
                         __bf16* __restrict__ hch, float* __restrict__ cch,
                         int* __restrict__ lengths, int lvl) {
  const int n = blockIdx.x;
  const int c = threadIdx.x;  // DD threads
  int len = 0;
  #pragma unroll
  for (int k = 0; k < KK; ++k) {
    int idx = indices[(lvl * NN + n) * KK + k];  // uniform -> scalar load
    len += (idx != -1) ? 1 : 0;
    float hv = 0.f, cv = 0.f;
    if (idx > 0) {  // row 0 of h_tab is the zero row; -1 invalid
      hv = prevh[(size_t)(idx - 1) * DD + c];
      cv = prevc[(size_t)(idx - 1) * DD + c];
    }
    hch[((size_t)n * KK + k) * DD + c] = f2bf(hv);
    cch[((size_t)n * KK + k) * DD + c] = cv;
  }
  if (c == 0) lengths[n] = len;
}

// ---------------------------------------------------------------------------
// Wx = x[l] @ W_w + W_b via WMMA; pre-packed bf16 operands.
// ---------------------------------------------------------------------------
__global__ void k_wx(const __bf16* __restrict__ xbf, const __bf16* __restrict__ wwpk,
                     const float* __restrict__ Wb, float* __restrict__ wx, int lvl) {
  const int lane = threadIdx.x;
  const int m = lane & 15, kh = lane >> 4;
  const int nb = blockIdx.x * 16;
  const int nt = blockIdx.y;
  const int col = nt * 16 + m;
  const __bf16* xrow = xbf + (size_t)(lvl * NN + nb + m) * DIN;
  v8f acc;
  float bias = Wb[col];
  #pragma unroll
  for (int r = 0; r < 8; ++r) acc[r] = bias;
  #pragma unroll
  for (int kt = 0; kt < 8; ++kt) {
    const int kb0 = kt * 32 + kh * 8, kb1 = kt * 32 + 16 + kh * 8;
    v16bf a = cat16(*(const v8bf*)(xrow + kb0), *(const v8bf*)(xrow + kb1));
    v16bf b = *(const v16bf*)(wwpk + ((((kt * 32 + nt) * 32) + lane) << 4));
    acc = wmma_bf(a, b, acc);
  }
  #pragma unroll
  for (int r = 0; r < 8; ++r) {
    int row = r + kh * 8;
    wx[(size_t)(nb + row) * G4 + col] = acc[r];
  }
}

// ---------------------------------------------------------------------------
// Fused masked BiLSTM recurrence + fc projection, one (gate,dir) per
// blockIdx.y. 2 waves/block, 16 nodes/wave. Weight fragments are DMA'd into
// LDS with async-to-LDS loads and consumed by v_wmma from there.
// ---------------------------------------------------------------------------
__global__ void k_bilstm(const __bf16* __restrict__ hch,
                         const int* __restrict__ lengths,
                         const __bf16* __restrict__ wpk,
                         const __bf16* __restrict__ fcpk,
                         const float* __restrict__ ubih,
                         const float* __restrict__ ubhh,
                         const float* __restrict__ uh0,
                         const float* __restrict__ uc0,
                         float* __restrict__ flin, float* __restrict__ glast) {
  extern __shared__ char smem[];
  __bf16* wfrag  = (__bf16*)smem;
  __bf16* fcfrag = wfrag + SM_WFRAG_ELEMS;
  __bf16* hstA   = fcfrag + SM_FCFRAG_ELEMS;
  __bf16* ylA    = hstA + SM_HST_ELEMS;
  int*    lenbuf = (int*)(ylA + SM_YL_ELEMS);

  const int tid = threadIdx.x;
  const int lane = tid & 31;
  const int wave = tid >> 5;
  const int gd = blockIdx.y;
  const int gate = gd >> 1, dir = gd & 1;

  const float* bih = ubih + (size_t)gd * G4;
  const float* bhh = ubhh + (size_t)gd * G4;

  // ---- async DMA of pre-packed fragments into LDS ----
  {
    const char* wsrc = (const char*)(wpk + (size_t)gd * SM_WFRAG_ELEMS);
    const uint32_t wdst = lds_off(wfrag);
    for (int b = tid * 16; b < SM_WFRAG_ELEMS * 2; b += (int)blockDim.x * 16)
      async_copy16(wdst + b, wsrc + b);
    const char* fsrc = (const char*)(fcpk + (size_t)gd * SM_FCFRAG_ELEMS);
    const uint32_t fdst = lds_off(fcfrag);
    for (int b = tid * 16; b < SM_FCFRAG_ELEMS * 2; b += (int)blockDim.x * 16)
      async_copy16(fdst + b, fsrc + b);
  }
  if (tid < 32) lenbuf[tid] = lengths[blockIdx.x * 32 + tid];
  asm volatile("s_wait_asynccnt 0" ::: "memory");
  __syncthreads();

  const int m = lane & 15;
  const int kh = lane >> 4;
  const int nodebase = blockIdx.x * 32 + wave * 16;
  const int node = nodebase + m;
  __bf16* hst = hstA + wave * 16 * DD;
  __bf16* yl  = ylA  + wave * 16 * DD;

  // init h state (bf16, row-major [16][128] == A-operand friendly)
  for (int i = lane; i < 16 * DD; i += 32)
    hst[i] = f2bf(uh0[gate * DD + (i & (DD - 1))]);

  // c state in registers: cst[j][r] covers col j*16+m, row r+8*kh
  float cst[8][8];
  #pragma unroll
  for (int j = 0; j < 8; ++j) {
    float c0 = uc0[gate * DD + j * 16 + m];
    #pragma unroll
    for (int r = 0; r < 8; ++r) cst[j][r] = c0;
  }
  float biasv[8][4];
  #pragma unroll
  for (int j = 0; j < 8; ++j)
    #pragma unroll
    for (int G = 0; G < 4; ++G) {
      int col = G * DD + j * 16 + m;
      biasv[j][G] = bih[col] + bhh[col];
    }
  int lenr[8];
  #pragma unroll
  for (int r = 0; r < 8; ++r) lenr[r] = lenbuf[wave * 16 + r + kh * 8];
  const int lenm = lenbuf[wave * 16 + m];

  for (int s = 0; s < KK; ++s) {
    const int t = dir ? (KK - 1 - s) : s;

    // A fragments: kt 0..3 from x_t, kt 4..7 from h state
    v16bf afr[8];
    const __bf16* xrow = hch + (size_t)(node * KK + t) * DD;
    #pragma unroll
    for (int kt = 0; kt < 4; ++kt) {
      const int kb0 = kt * 32 + kh * 8, kb1 = kt * 32 + 16 + kh * 8;
      afr[kt] = cat16(*(const v8bf*)(xrow + kb0), *(const v8bf*)(xrow + kb1));
    }
    asm volatile("s_wait_dscnt 0" ::: "memory");  // h-state RAW (prev step)
    #pragma unroll
    for (int kt = 0; kt < 4; ++kt) {
      const int kb0 = kt * 32 + kh * 8, kb1 = kt * 32 + 16 + kh * 8;
      afr[4 + kt] = cat16(*(const v8bf*)(hst + m * DD + kb0),
                          *(const v8bf*)(hst + m * DD + kb1));
    }

    #pragma unroll
    for (int j = 0; j < 8; ++j) {
      v8f acc[4];
      #pragma unroll
      for (int G = 0; G < 4; ++G)
        #pragma unroll
        for (int r = 0; r < 8; ++r) acc[G][r] = biasv[j][G];
      #pragma unroll
      for (int G = 0; G < 4; ++G) {          // torch gate order i,f,g,o
        const int nt = G * 8 + j;
        #pragma unroll
        for (int kt = 0; kt < 8; ++kt) {
          v16bf b = *(const v16bf*)(wfrag + ((((kt * 32 + nt) * 32) + lane) << 4));
          acc[G] = wmma_bf(afr[kt], b, acc[G]);
        }
      }
      // fused masked LSTM cell update
      #pragma unroll
      for (int r = 0; r < 8; ++r) {
        const int row = r + kh * 8;
        const bool valid = t < lenr[r];
        float iv = sigmoidf_(acc[0][r]);
        float fv = sigmoidf_(acc[1][r]);
        float gv = tanhf_(acc[2][r]);
        float ov = sigmoidf_(acc[3][r]);
        float cn = fv * cst[j][r] + iv * gv;
        float hn = ov * tanhf_(cn);
        if (valid) {
          cst[j][r] = cn;
          hst[row * DD + j * 16 + m] = f2bf(hn);
          if (gate != 0 && (lenr[r] - 1 == t))
            yl[row * DD + j * 16 + m] = f2bf(hn);
        }
      }
    }

    if (gate == 0) {  // per-step fc partial: y_t @ fc_half (masked rows -> 0)
      asm volatile("s_wait_dscnt 0" ::: "memory");
      const bool mvalid = t < lenm;
      v16bf ay[4];
      #pragma unroll
      for (int kt = 0; kt < 4; ++kt) {
        const int kb0 = kt * 32 + kh * 8, kb1 = kt * 32 + 16 + kh * 8;
        v16bf a = cat16(*(const v8bf*)(hst + m * DD + kb0),
                        *(const v8bf*)(hst + m * DD + kb1));
        #pragma unroll
        for (int e = 0; e < 16; ++e) a[e] = mvalid ? a[e] : f2bf(0.f);
        ay[kt] = a;
      }
      #pragma unroll
      for (int nt = 0; nt < 8; ++nt) {
        v8f acc;
        #pragma unroll
        for (int r = 0; r < 8; ++r) acc[r] = 0.f;
        #pragma unroll
        for (int kt = 0; kt < 4; ++kt) {
          v16bf b = *(const v16bf*)(fcfrag + ((((kt * 8 + nt) * 32) + lane) << 4));
          acc = wmma_bf(ay[kt], b, acc);
        }
        #pragma unroll
        for (int r = 0; r < 8; ++r) {
          const int row = r + kh * 8;
          const int n = nodebase + row;
          flin[((size_t)(dir * NN + n) * KK + t) * DD + nt * 16 + m] = acc[r];
        }
      }
    }
  }

  if (gate != 0) {  // y_last @ fc_half
    asm volatile("s_wait_dscnt 0" ::: "memory");
    v16bf ay[4];
    #pragma unroll
    for (int kt = 0; kt < 4; ++kt) {
      const int kb0 = kt * 32 + kh * 8, kb1 = kt * 32 + 16 + kh * 8;
      ay[kt] = cat16(*(const v8bf*)(yl + m * DD + kb0),
                     *(const v8bf*)(yl + m * DD + kb1));
    }
    #pragma unroll
    for (int nt = 0; nt < 8; ++nt) {
      v8f acc;
      #pragma unroll
      for (int r = 0; r < 8; ++r) acc[r] = 0.f;
      #pragma unroll
      for (int kt = 0; kt < 4; ++kt) {
        v16bf b = *(const v16bf*)(fcfrag + ((((kt * 8 + nt) * 32) + lane) << 4));
        acc = wmma_bf(ay[kt], b, acc);
      }
      #pragma unroll
      for (int r = 0; r < 8; ++r) {
        const int row = r + kh * 8;
        const int n = nodebase + row;
        glast[((size_t)((gate - 1) * 2 + dir) * NN + n) * DD + nt * 16 + m] = acc[r];
      }
    }
  }
}

// ---------------------------------------------------------------------------
// Tree-cell combine + output + next-level tables.
// ---------------------------------------------------------------------------
__global__ void k_combine(const float* __restrict__ wx,
                          const float* __restrict__ flin,
                          const float* __restrict__ glast,
                          const float* __restrict__ cch,
                          const int* __restrict__ lengths,
                          float* __restrict__ out, float* __restrict__ tabh,
                          float* __restrict__ tabc, int lvl) {
  int i = blockIdx.x * blockDim.x + threadIdx.x;  // NN*DD
  if (i >= NN * DD) return;
  int c = i & (DD - 1);
  int n = i >> 7;
  int len = lengths[n];
  const float* wxr = wx + (size_t)n * G4;
  float Wf = wxr[0 * DD + c], Wi = wxr[1 * DD + c];
  float Wu = wxr[2 * DD + c], Wo = wxr[3 * DD + c];
  float branch_f = 0.f;
  for (int k = 0; k < len; ++k) {
    float fl = flin[((size_t)(0 * NN + n) * KK + k) * DD + c] +
               flin[((size_t)(1 * NN + n) * KK + k) * DD + c];
    branch_f += sigmoidf_(Wf + fl) * cch[((size_t)n * KK + k) * DD + c];
  }
  float gi = glast[((size_t)(0) * NN + n) * DD + c] + glast[((size_t)(1) * NN + n) * DD + c];
  float gu = glast[((size_t)(2) * NN + n) * DD + c] + glast[((size_t)(3) * NN + n) * DD + c];
  float go = glast[((size_t)(4) * NN + n) * DD + c] + glast[((size_t)(5) * NN + n) * DD + c];
  float iv = sigmoidf_(Wi + gi);
  float uv = tanhf_(Wu + gu);
  float ov = sigmoidf_(Wo + go);
  float nc = iv * uv + branch_f;
  float nh = ov * tanhf_(nc);
  out[((size_t)(0 * LVLS + lvl) * NN + n) * DD + c] = nh;
  out[((size_t)(1 * LVLS + lvl) * NN + n) * DD + c] = nc;
  tabh[i] = nh;
  tabc[i] = nc;
}

// ---------------------------------------------------------------------------
extern "C" void kernel_launch(void* const* d_in, const int* in_sizes, int n_in,
                              void* d_out, int out_size, void* d_ws, size_t ws_size,
                              hipStream_t stream) {
  const float* x       = (const float*)d_in[0];
  const int*   indices = (const int*)d_in[1];
  const float* Ww      = (const float*)d_in[2];
  const float* Wb      = (const float*)d_in[3];
  const float* uh0     = (const float*)d_in[4];
  const float* uc0     = (const float*)d_in[5];
  const float* uWih    = (const float*)d_in[6];
  const float* uWhh    = (const float*)d_in[7];
  const float* ubih    = (const float*)d_in[8];
  const float* ubhh    = (const float*)d_in[9];
  const float* ufc     = (const float*)d_in[10];
  float* out = (float*)d_out;

  char* ws = (char*)d_ws;
  size_t off = 0;
  auto alloc = [&](size_t bytes) {
    char* p = ws + off;
    off += (bytes + 255) & ~(size_t)255;
    return p;
  };
  float*  wx      = (float*)alloc(sizeof(float) * NN * G4);
  __bf16* hch     = (__bf16*)alloc((size_t)2 * NN * KK * DD);
  float*  cch     = (float*)alloc(sizeof(float) * NN * KK * DD);
  int*    lengths = (int*)alloc(sizeof(int) * NN);
  float*  flin    = (float*)alloc(sizeof(float) * 2 * NN * KK * DD);
  float*  glast   = (float*)alloc(sizeof(float) * 6 * NN * DD);
  float*  tabh    = (float*)alloc(sizeof(float) * NN * DD);
  float*  tabc    = (float*)alloc(sizeof(float) * NN * DD);
  __bf16* wpk     = (__bf16*)alloc((size_t)2 * 8 * SM_WFRAG_ELEMS);
  __bf16* fcpk    = (__bf16*)alloc((size_t)2 * 8 * SM_FCFRAG_ELEMS);
  __bf16* wwpk    = (__bf16*)alloc((size_t)2 * SM_WFRAG_ELEMS);
  __bf16* xbf     = (__bf16*)alloc((size_t)2 * LVLS * NN * DIN);
  (void)in_sizes; (void)n_in; (void)out_size; (void)ws_size;

  const size_t smemK2 =
      (size_t)2 * (SM_WFRAG_ELEMS + SM_FCFRAG_ELEMS + SM_HST_ELEMS + SM_YL_ELEMS) +
      sizeof(int) * 32;

  // one-time packs (deterministic, every call)
  k_pack_wcat<<<dim3(8 * SM_WFRAG_ELEMS / 256), dim3(256), 0, stream>>>(uWih, uWhh, wpk);
  k_pack_fc<<<dim3(8 * SM_FCFRAG_ELEMS / 256), dim3(256), 0, stream>>>(ufc, fcpk);
  k_pack_ww<<<dim3(SM_WFRAG_ELEMS / 256), dim3(256), 0, stream>>>(Ww, wwpk);
  k_pack_x<<<dim3(LVLS * NN * DIN / 256), dim3(256), 0, stream>>>(x, xbf);

  for (int l = 0; l < LVLS; ++l) {
    k_gather<<<dim3(NN), dim3(DD), 0, stream>>>(indices, tabh, tabc, hch, cch,
                                                lengths, l);
    k_wx<<<dim3(NN / 16, G4 / 16), dim3(32), 0, stream>>>(xbf, wwpk, Wb, wx, l);
    k_bilstm<<<dim3(NN / 32, 8), dim3(64), smemK2, stream>>>(
        hch, lengths, wpk, fcpk, ubih, ubhh, uh0, uc0, flin, glast);
    k_combine<<<dim3((NN * DD + 255) / 256), dim3(256), 0, stream>>>(
        wx, flin, glast, cch, lengths, out, tabh, tabc, l);
  }
}
